// CrossModalAttention_58282706206902
// MI455X (gfx1250) — matrix operand order repaired and checked by reading, not compiled, not run on hardware
//
#include <hip/hip_runtime.h>
#include <hip/hip_bf16.h>

typedef __bf16 bf16_t;
typedef __attribute__((ext_vector_type(16))) __bf16 v16bf;
typedef __attribute__((ext_vector_type(8)))  float  v8f;
typedef unsigned int u32x4 __attribute__((ext_vector_type(4)));
typedef int          i32x4 __attribute__((ext_vector_type(4)));
typedef int          i32x8 __attribute__((ext_vector_type(8)));

#define N_TOK 4096
#define CT    256
#define DHEAD 32
#define NHEAD 8
#define HW    64

// Padded LDS row lengths (halves): chosen so 16 lanes' b128 reads hit 16
// distinct bank groups (20*hl mod 64 / 36*hl mod 64 are all distinct).
#define LD40 40   // 80B rows  (32-half payload + 16B pad)
#define LD72 72   // 144B rows (64-half payload + 16B pad)

#if defined(__has_builtin)
#if __has_builtin(__builtin_amdgcn_tensor_load_to_lds) && \
    __has_builtin(__builtin_amdgcn_s_wait_tensorcnt)
#define HAVE_TDM 1
#endif
#endif
#ifndef HAVE_TDM
#define HAVE_TDM 0
#endif

// ---------------------------------------------------------------------------
// WMMA fragment loaders (wave32, contiguous k = 16*hi + t in both A and B), expressed as
// two contiguous 16B chunks per lane -> ds_load_b128 / global_load_b128.
// ---------------------------------------------------------------------------

// A-matrix 16x32 bf16 from row-major [M][K] source, row stride ld (halves).
// half t = K (hi?16:0)+t ; M=hl.
__device__ __forceinline__ v16bf load_A_frag(const bf16_t* base, int ld) {
  const int lane = threadIdx.x & 31;
  const int hl = lane & 15, hi = lane >> 4;
  union { uint4 q[2]; v16bf v; } u;
  u.q[0] = *(const uint4*)(base + hl * ld + (hi ? 16 : 0));
  u.q[1] = *(const uint4*)(base + hl * ld + (hi ? 24 : 8));
  return u.v;
}

// B-matrix 32x16 bf16 from an [N][K]-major source (i.e. the transpose of B),
// row stride ld (halves). half t = K (hi?16:0)+t, N = hl -> one contiguous
// 32B run per lane.
__device__ __forceinline__ v16bf load_B_fragT(const bf16_t* base, int ld) {
  const int lane = threadIdx.x & 31;
  const int hl = lane & 15, hi = lane >> 4;
  union { uint4 q[2]; v16bf v; } u;
  const uint4* p = (const uint4*)(base + hl * ld + hi * 16);
  u.q[0] = p[0];
  u.q[1] = p[1];
  return u.v;
}

#if HAVE_TDM
// ---------------------------------------------------------------------------
// Tensor Data Mover: DMA one 2D tile (row-major, 2-byte elements) from global
// into LDS, hardware-padding each tile row so LDS rows land on the padded
// (bank-conflict-free) pitches used by the fragment loaders.
//   pad_interval: log2(DWORDs between pads)-1  (3 -> every 16 DW = 64B row)
//   pad_amount  : DWORDs of pad minus 1        (3 -> 4 DW = 16B pad)
// Descriptor packing per CDNA5 ISA 8.3/8.4 (D# group0/group1).
// ---------------------------------------------------------------------------
__device__ __forceinline__ void tdm_load_tile_2d(const bf16_t* gptr,
                                                 unsigned lds_off,
                                                 unsigned tensor_d0,
                                                 unsigned tensor_d1,
                                                 unsigned tile_d0,
                                                 unsigned tile_d1,
                                                 unsigned stride0,
                                                 unsigned pad_interval,
                                                 unsigned pad_amount) {
  const unsigned long long ga = (unsigned long long)(size_t)gptr;
  u32x4 g0;
  g0[0] = 1u;                                         // count=1, user descriptor
  g0[1] = lds_off;                                    // lds_addr (bytes)
  g0[2] = (unsigned)(ga & 0xffffffffu);               // global_addr[31:0]
  g0[3] = (unsigned)((ga >> 32) & 0x01ffffffu)        // global_addr[56:32]
          | 0x80000000u;                              // type = 2 ("image")
  i32x8 g1;
  g1[0] = (int)((1u << 16)                            // data_size = 2B
                | (1u << 20)                          // pad_enable
                | (pad_interval << 22) | (pad_amount << 25));
  g1[1] = (int)((tensor_d0 & 0xffffu) << 16);         // dim0[15:0] (abar=0)
  g1[2] = (int)((tensor_d0 >> 16) | ((tensor_d1 & 0xffffu) << 16));
  g1[3] = (int)((tensor_d1 >> 16) | (tile_d0 << 16)); // dim1 hi | tile_dim0
  g1[4] = (int)(tile_d1 & 0xffffu);                   // tile_dim1 (tile_dim2=0)
  g1[5] = (int)stride0;                               // dim0_stride[31:0]
  g1[6] = 0;
  g1[7] = 0;
  const i32x4 z4 = {0, 0, 0, 0};
#if defined(__clang_major__) && __clang_major__ >= 23
  const i32x8 z8 = {0, 0, 0, 0, 0, 0, 0, 0};
  __builtin_amdgcn_tensor_load_to_lds(g0, g1, z4, z4, z8, 0);
#else
  __builtin_amdgcn_tensor_load_to_lds(g0, g1, z4, z4, 0);
#endif
}
#endif  // HAVE_TDM

// ---------------------------------------------------------------------------
// Emissivity gate: Conv(1->32,3x3,SAME) -> ReLU -> Conv(32->8,1x1) -> Sigmoid
// g[h][n] gates query rows; folded into softmax row scale downstream.
// ---------------------------------------------------------------------------
__global__ void gate_kernel(const float* __restrict__ emis,
                            const float* __restrict__ w1, const float* __restrict__ b1,
                            const float* __restrict__ w2, const float* __restrict__ b2,
                            float* __restrict__ g) {
  const int n = blockIdx.x * blockDim.x + threadIdx.x;
  if (n >= N_TOK) return;
  const int y = n >> 6, x = n & 63;
  float pix[9];
#pragma unroll
  for (int dy = -1; dy <= 1; ++dy)
#pragma unroll
    for (int dx = -1; dx <= 1; ++dx) {
      const int yy = y + dy, xx = x + dx;
      pix[(dy + 1) * 3 + (dx + 1)] =
          (yy >= 0 && yy < HW && xx >= 0 && xx < HW) ? emis[yy * HW + xx] : 0.f;
    }
  float acc[NHEAD];
#pragma unroll
  for (int h = 0; h < NHEAD; ++h) acc[h] = b2[h];
  for (int c = 0; c < 32; ++c) {
    float s = b1[c];
#pragma unroll
    for (int t = 0; t < 9; ++t) s += w1[c * 9 + t] * pix[t];
    s = fmaxf(s, 0.f);
#pragma unroll
    for (int h = 0; h < NHEAD; ++h) acc[h] += w2[h * 32 + c] * s;
  }
#pragma unroll
  for (int h = 0; h < NHEAD; ++h)
    g[h * N_TOK + n] = 1.f / (1.f + __expf(-acc[h]));
}

// ---------------------------------------------------------------------------
// Generic 64x64-tile bf16 WMMA GEMM:  out[n][c] = A[n][:] . W[c][:] + bias[c]
//   a_trans==1 : A[n][k] = Asrc[k*N_TOK + n]   (channel-major activations)
//   a_trans==0 : A[n][k] = Asrc[n*Cin + k]     (row-major)
//   out_mode 0 : bf16, per-head layout [c/32][n][c%32]        (Q,K buffers)
//   out_mode 2 : bf16, per-head transposed [c/32][c%32][n]    (V buffer)
//   out_mode 1 : f32, transposed layout  out[c*N_TOK + n]     (final output)
// ---------------------------------------------------------------------------
__global__ void __launch_bounds__(128)
proj_gemm_kernel(const float* __restrict__ Asrc, const float* __restrict__ W,
                 const float* __restrict__ bias, int Cin, int a_trans,
                 int out_mode, bf16_t* __restrict__ out_bf,
                 float* __restrict__ out_f) {
  __shared__ alignas(16) bf16_t sA[64][LD40];
  __shared__ alignas(16) bf16_t sBT[64][LD40];
  const int tid = threadIdx.x;
  const int wave = tid >> 5;
  const int lane = tid & 31, hl = lane & 15, hi = lane >> 4;
  const int m0 = blockIdx.x * 64;
  const int c0 = blockIdx.y * 64;

  v8f acc[4];
#pragma unroll
  for (int nb = 0; nb < 4; ++nb)
#pragma unroll
    for (int r = 0; r < 8; ++r) acc[nb][r] = 0.f;

  for (int k0 = 0; k0 < Cin; k0 += 32) {
    __syncthreads();
    if (a_trans) {
      // consecutive lanes -> consecutive n at fixed k : coalesced global reads
#pragma unroll
      for (int i = tid; i < 64 * 32; i += 128) {
        const int kk = i >> 6, m = i & 63;
        sA[m][kk] = (bf16_t)Asrc[(size_t)(k0 + kk) * N_TOK + (m0 + m)];
      }
    } else {
#pragma unroll
      for (int i = tid; i < 64 * 32; i += 128) {
        const int m = i >> 5, kk = i & 31;
        sA[m][kk] = (bf16_t)Asrc[(size_t)(m0 + m) * Cin + (k0 + kk)];
      }
    }
#pragma unroll
    for (int i = tid; i < 64 * 32; i += 128) {
      const int c = i >> 5, kk = i & 31;
      sBT[c][kk] = (bf16_t)W[(size_t)(c0 + c) * Cin + (k0 + kk)];
    }
    __syncthreads();
    const v16bf a = load_A_frag(&sA[wave * 16][0], LD40);
#pragma unroll
    for (int nb = 0; nb < 4; ++nb) {
      const v16bf b = load_B_fragT(&sBT[nb * 16][0], LD40);
      acc[nb] = __builtin_amdgcn_wmma_f32_16x16x32_bf16(
          false, a, false, b, (short)0, acc[nb], false, false);
    }
  }

#pragma unroll
  for (int nb = 0; nb < 4; ++nb)
#pragma unroll
    for (int r = 0; r < 8; ++r) {
      const int n = m0 + wave * 16 + (hi ? 8 : 0) + r;
      const int c = c0 + nb * 16 + hl;
      const float val = acc[nb][r] + bias[c];
      if (out_mode == 1) {
        out_f[(size_t)c * N_TOK + n] = val;
      } else if (out_mode == 2) {  // [h][d][n] : V stored pre-transposed
        out_bf[((size_t)(c >> 5) * DHEAD + (c & 31)) * N_TOK + n] = (bf16_t)val;
      } else {                     // [h][n][d]
        out_bf[((size_t)(c >> 5) * N_TOK + n) * DHEAD + (c & 31)] = (bf16_t)val;
      }
    }
}

// ---------------------------------------------------------------------------
// Flash attention: one workgroup (4 waves) per (head, 64-query tile).
// Q fragment pinned in registers; 64-key K/V tiles double-buffered in LDS,
// streamed by the Tensor Data Mover (hardware row padding) one tile ahead of
// compute. V arrives pre-transposed [d][key] so every LDS fragment read is a
// contiguous pair of ds_load_b128.
// ---------------------------------------------------------------------------
__global__ void __launch_bounds__(128)
flash_attn_kernel(const bf16_t* __restrict__ qbuf, const bf16_t* __restrict__ kbuf,
                  const bf16_t* __restrict__ vtb, const float* __restrict__ g,
                  float* __restrict__ Obuf) {
  __shared__ alignas(16) bf16_t sK[2][64][LD40];      // [key][d]
  __shared__ alignas(16) bf16_t sVT[2][DHEAD][LD72];  // [d][key]
  __shared__ alignas(16) bf16_t sP[4][16][LD72];      // per-wave P tile [m][key]
  const int tid = threadIdx.x;
  const int wave = tid >> 5;
  const int lane = tid & 31, hl = lane & 15, hi = lane >> 4;
  const int head = blockIdx.y;
  const int myq0 = blockIdx.x * 64 + wave * 16;
  const float scale = 0.17677669529663687f;  // 1/sqrt(32)
  const int NT = N_TOK / 64;

  const size_t hoff = (size_t)head * N_TOK * DHEAD;
  const v16bf qa = load_A_frag(qbuf + hoff + (size_t)myq0 * DHEAD, DHEAD);

  float gs[8], m_r[8], l_r[8];
#pragma unroll
  for (int r = 0; r < 8; ++r) {
    gs[r] = scale * g[head * N_TOK + myq0 + (hi ? 8 : 0) + r];
    m_r[r] = -1e30f;
    l_r[r] = 0.f;
  }
  v8f o0, o1;
#pragma unroll
  for (int r = 0; r < 8; ++r) { o0[r] = 0.f; o1[r] = 0.f; }

#if HAVE_TDM
  if (wave == 0) {  // DMA tile 0 into buffer 0 (issued once per workgroup)
    tdm_load_tile_2d(kbuf + hoff, (unsigned)(size_t)&sK[0][0][0],
                     DHEAD, N_TOK, DHEAD, 64, DHEAD, 3, 3);
    tdm_load_tile_2d(vtb + hoff, (unsigned)(size_t)&sVT[0][0][0],
                     N_TOK, DHEAD, 64, DHEAD, N_TOK, 4, 3);
  }
#endif

  for (int kt = 0; kt < NT; ++kt) {
    const int cur = kt & 1;
    const int k0 = kt * 64;
#if HAVE_TDM
    if (wave == 0) __builtin_amdgcn_s_wait_tensorcnt(0);
    __syncthreads();  // tile kt visible to all; tile kt-1 compute finished
    if (wave == 0 && kt + 1 < NT) {  // DMA tile kt+1 behind the compute
      const int kn = k0 + 64;
      tdm_load_tile_2d(kbuf + hoff + (size_t)kn * DHEAD,
                       (unsigned)(size_t)&sK[cur ^ 1][0][0],
                       DHEAD, N_TOK, DHEAD, 64, DHEAD, 3, 3);
      tdm_load_tile_2d(vtb + hoff + kn,
                       (unsigned)(size_t)&sVT[cur ^ 1][0][0],
                       N_TOK, DHEAD, 64, DHEAD, N_TOK, 4, 3);
    }
#else
    __syncthreads();
    {  // manual fallback: 256 uint4 per matrix, coalesced b128 both sides
#pragma unroll
      for (int it = 0; it < 2; ++it) {
        const int i = tid + it * 128;
        const int key = i >> 2, db = (i & 3) * 8;
        *(uint4*)&sK[cur][key][db] =
            *(const uint4*)(kbuf + hoff + (size_t)(k0 + key) * DHEAD + db);
        const int d = i >> 3, kc = (i & 7) * 8;
        *(uint4*)&sVT[cur][d][kc] =
            *(const uint4*)(vtb + hoff + (size_t)d * N_TOK + k0 + kc);
      }
    }
    if (kt + 1 < NT) {
      __builtin_prefetch(kbuf + hoff + (size_t)(k0 + 64) * DHEAD + tid * 16, 0, 1);
      __builtin_prefetch(vtb + hoff + (size_t)(k0 + 64) + tid * 4, 0, 1);
    }
    __syncthreads();
#endif

    // S block (16 rows x 64 keys): B = K^T, fragments are K-tile row reads.
    v8f s[4];
#pragma unroll
    for (int nb = 0; nb < 4; ++nb) {
      const v16bf b = load_B_fragT(&sK[cur][nb * 16][0], LD40);
      v8f zc;
#pragma unroll
      for (int r = 0; r < 8; ++r) zc[r] = 0.f;
      s[nb] = __builtin_amdgcn_wmma_f32_16x16x32_bf16(
          false, qa, false, b, (short)0, zc, false, false);
    }

    // online softmax; rows for (r,hi) span lanes hl=0..15 -> xor-reduce 1,2,4,8
#pragma unroll
    for (int r = 0; r < 8; ++r) {
      float mx = -1e30f;
#pragma unroll
      for (int nb = 0; nb < 4; ++nb) {
        s[nb][r] *= gs[r];
        mx = fmaxf(mx, s[nb][r]);
      }
#pragma unroll
      for (int msk = 1; msk < 16; msk <<= 1)
        mx = fmaxf(mx, __shfl_xor(mx, msk, 32));
      const float mnew = fmaxf(m_r[r], mx);
      const float corr = __expf(m_r[r] - mnew);
      float rsum = 0.f;
#pragma unroll
      for (int nb = 0; nb < 4; ++nb) {
        const float p = __expf(s[nb][r] - mnew);
        rsum += p;
        sP[wave][(hi ? 8 : 0) + r][nb * 16 + hl] = (bf16_t)p;
      }
#pragma unroll
      for (int msk = 1; msk < 16; msk <<= 1)
        rsum += __shfl_xor(rsum, msk, 32);
      l_r[r] = l_r[r] * corr + rsum;
      m_r[r] = mnew;
      o0[r] *= corr;
      o1[r] *= corr;
    }

    // O += P (16x64) x V (64x32): two K=32 steps, two N-blocks of 16.
#pragma unroll
    for (int kb = 0; kb < 2; ++kb) {
      const v16bf pa = load_A_frag(&sP[wave][0][kb * 32], LD72);
      const v16bf b0 = load_B_fragT(&sVT[cur][0][kb * 32], LD72);
      o0 = __builtin_amdgcn_wmma_f32_16x16x32_bf16(
          false, pa, false, b0, (short)0, o0, false, false);
      const v16bf b1 = load_B_fragT(&sVT[cur][16][kb * 32], LD72);
      o1 = __builtin_amdgcn_wmma_f32_16x16x32_bf16(
          false, pa, false, b1, (short)0, o1, false, false);
    }
  }

  // normalize and write O in [n][256] row-major (feeds final projection GEMM)
#pragma unroll
  for (int r = 0; r < 8; ++r) {
    const float inv = 1.0f / l_r[r];
    const int n = myq0 + (hi ? 8 : 0) + r;
    Obuf[(size_t)n * CT + head * DHEAD + hl]      = o0[r] * inv;
    Obuf[(size_t)n * CT + head * DHEAD + 16 + hl] = o1[r] * inv;
  }
}

// ---------------------------------------------------------------------------
extern "C" void kernel_launch(void* const* d_in, const int* in_sizes, int n_in,
                              void* d_out, int out_size, void* d_ws, size_t ws_size,
                              hipStream_t stream) {
  (void)in_sizes; (void)n_in; (void)out_size; (void)ws_size;
  const float* thermal = (const float*)d_in[0];
  const float* optical = (const float*)d_in[1];
  const float* emis    = (const float*)d_in[2];
  const float* Wq = (const float*)d_in[3];
  const float* bq = (const float*)d_in[4];
  const float* Wk = (const float*)d_in[5];
  const float* bk = (const float*)d_in[6];
  const float* Wv = (const float*)d_in[7];
  const float* bv = (const float*)d_in[8];
  const float* c1w = (const float*)d_in[9];
  const float* c1b = (const float*)d_in[10];
  const float* c2w = (const float*)d_in[11];
  const float* c2b = (const float*)d_in[12];
  const float* Wp = (const float*)d_in[13];
  const float* bp = (const float*)d_in[14];
  float* out = (float*)d_out;

  char* ws = (char*)d_ws;
  size_t off = 0;
  float*  gbuf = (float*)(ws + off);  off += (size_t)NHEAD * N_TOK * sizeof(float);
  bf16_t* qbuf = (bf16_t*)(ws + off); off += (size_t)N_TOK * CT * sizeof(bf16_t);
  bf16_t* kbuf = (bf16_t*)(ws + off); off += (size_t)N_TOK * CT * sizeof(bf16_t);
  bf16_t* vtb  = (bf16_t*)(ws + off); off += (size_t)N_TOK * CT * sizeof(bf16_t);
  float*  Obuf = (float*)(ws + off);  off += (size_t)N_TOK * CT * sizeof(float);

  gate_kernel<<<N_TOK / 128, 128, 0, stream>>>(emis, c1w, c1b, c2w, c2b, gbuf);

  dim3 ggrid(N_TOK / 64, CT / 64);
  proj_gemm_kernel<<<ggrid, 128, 0, stream>>>(thermal, Wq, bq, CT, 1, 0, qbuf, nullptr);
  proj_gemm_kernel<<<ggrid, 128, 0, stream>>>(optical, Wk, bk, 128, 1, 0, kbuf, nullptr);
  proj_gemm_kernel<<<ggrid, 128, 0, stream>>>(optical, Wv, bv, 128, 1, 2, vtb, nullptr);

  flash_attn_kernel<<<dim3(N_TOK / 64, NHEAD), 128, 0, stream>>>(qbuf, kbuf, vtb,
                                                                 gbuf, Obuf);

  proj_gemm_kernel<<<ggrid, 128, 0, stream>>>(Obuf, Wp, bp, CT, 0, 1, nullptr, out);
}
